// G_Block_47734266528082
// MI455X (gfx1250) — matrix operand order repaired and checked by reading, not compiled, not run on hardware
//
#include <hip/hip_runtime.h>

#define NB   8
#define CIN  256
#define CINT 192
#define COUT 128
#define LDIM 512
#define HH   128
#define WW   128
#define H2   256
#define W2   256
#define IMGC 3
#define EPSV 1e-8f

typedef __attribute__((ext_vector_type(8)))  _Float16 v8h;
typedef __attribute__((ext_vector_type(16))) _Float16 v16h;
typedef __attribute__((ext_vector_type(8)))  float    v8f;

typedef __attribute__((address_space(3))) _Float16 lds_f16_t;

__device__ __forceinline__ float lrelu(float v) { return v > 0.f ? v : 0.2f * v; }

// CDNA5 async copy: 16B global -> LDS, tracked by ASYNCcnt (inline asm: portable
// across ROCm-7.2 / amdgpu-toolchain builtin arity differences).
__device__ __forceinline__ void async_g2l_b128(_Float16* lds_dst, const _Float16* gsrc)
{
    uint32_t loff = (uint32_t)(uint64_t)(lds_f16_t*)lds_dst;  // LDS byte offset
    uint64_t ga   = (uint64_t)gsrc;
    asm volatile("global_load_async_to_lds_b128 %0, %1, off"
                 :: "v"(loff), "v"(ga)
                 : "memory");
}
__device__ __forceinline__ void wait_async_all()
{
    asm volatile("s_wait_asynccnt 0x0" ::: "memory");
}

// ---------------------------------------------------------------------------
// Kernel 1: styles + demod sigmas + modulated RGB weights (tiny)
// ---------------------------------------------------------------------------
__global__ void styles_kernel(const float* __restrict__ v,
                              const float* __restrict__ sw_up, const float* __restrict__ sb_up,
                              const float* __restrict__ w_up,
                              const float* __restrict__ sw_conv, const float* __restrict__ sb_conv,
                              const float* __restrict__ w_conv,
                              const float* __restrict__ sw_rgb, const float* __restrict__ sb_rgb,
                              const float* __restrict__ w_rgb,
                              float* __restrict__ s_up, float* __restrict__ sig_up,
                              float* __restrict__ s_conv, float* __restrict__ sig_conv,
                              float* __restrict__ s_rgb, float* __restrict__ wrgb)
{
    const int n = blockIdx.x;
    const int t = threadIdx.x;
    __shared__ float sh[CIN];
    const float* vn = v + n * LDIM;

    { // s_up[n][t], t < 256
        float acc = 0.f;
        const float* r = sw_up + t * LDIM;
        for (int l = 0; l < LDIM; ++l) acc += vn[l] * r[l];
        acc += sb_up[t] + 1.f;
        sh[t] = acc;
        s_up[n * CIN + t] = acc;
    }
    __syncthreads();
    if (t < CINT) { // sigma_up[n][t]
        float acc = EPSV;
        for (int c = 0; c < CIN; ++c) {
            const float* wp = w_up + (c * CINT + t) * 9;
            float wsq = 0.f;
            for (int k = 0; k < 9; ++k) wsq += wp[k] * wp[k];
            acc += sh[c] * sh[c] * wsq;
        }
        sig_up[n * CINT + t] = sqrtf(acc);
    }
    __syncthreads();
    if (t < CINT) { // s_conv
        float acc = 0.f;
        const float* r = sw_conv + t * LDIM;
        for (int l = 0; l < LDIM; ++l) acc += vn[l] * r[l];
        acc += sb_conv[t] + 1.f;
        sh[t] = acc;
        s_conv[n * CINT + t] = acc;
    }
    __syncthreads();
    if (t < COUT) {
        float acc = EPSV; // sigma_conv
        for (int c = 0; c < CINT; ++c) {
            const float* wp = w_conv + (t * CINT + c) * 9;
            float wsq = 0.f;
            for (int k = 0; k < 9; ++k) wsq += wp[k] * wp[k];
            acc += sh[c] * sh[c] * wsq;
        }
        sig_conv[n * COUT + t] = sqrtf(acc);
        float sr = 0.f; // s_rgb + modulated rgb weight (no demod)
        const float* r = sw_rgb + t * LDIM;
        for (int l = 0; l < LDIM; ++l) sr += vn[l] * r[l];
        sr += sb_rgb[t] + 1.f;
        s_rgb[n * COUT + t] = sr;
        for (int im = 0; im < IMGC; ++im)
            wrgb[(n * IMGC + im) * COUT + t] = w_rgb[im * COUT + t] * sr;
    }
}

// ---------------------------------------------------------------------------
// Kernel 2: fold style/demod into per-batch f16 weights.
// w1h: [n][ky][kx][o:CINT][ci:CIN],  w_eq[o,ci,ky,kx] = w_up[ci,o,2-ky,2-kx]
// ---------------------------------------------------------------------------
__global__ void modw_up_kernel(const float* __restrict__ w_up,
                               const float* __restrict__ s_up, const float* __restrict__ sig_up,
                               _Float16* __restrict__ w1h)
{
    int idx = blockIdx.x * blockDim.x + threadIdx.x;
    if (idx >= NB * 9 * CINT * CIN) return;
    int ci = idx % CIN; int r = idx / CIN;
    int o = r % CINT;   r /= CINT;
    int tap = r % 9;    int n = r / 9;
    int ky = tap / 3, kx = tap % 3;
    float w = w_up[(ci * CINT + o) * 9 + (2 - ky) * 3 + (2 - kx)];
    w1h[idx] = (_Float16)(w * s_up[n * CIN + ci] / sig_up[n * CINT + o]);
}

// w2h: [n][tap][o:COUT][ci:CINT]
__global__ void modw_conv_kernel(const float* __restrict__ w_conv,
                                 const float* __restrict__ s_conv, const float* __restrict__ sig_conv,
                                 _Float16* __restrict__ w2h)
{
    int idx = blockIdx.x * blockDim.x + threadIdx.x;
    if (idx >= NB * 9 * COUT * CINT) return;
    int ci = idx % CINT; int r = idx / CINT;
    int o = r % COUT;    r /= COUT;
    int tap = r % 9;     int n = r / 9;
    float w = w_conv[(o * CINT + ci) * 9 + tap];
    w2h[idx] = (_Float16)(w * s_conv[n * CINT + ci] / sig_conv[n * COUT + o]);
}

// ---------------------------------------------------------------------------
// Kernel 3: stride-2 transposed conv as 4 parity-class implicit GEMMs.
// Block: 32 out-ch x 256 parity-space pixels (16x16), 8 waves, 2x2 blocking.
// ---------------------------------------------------------------------------
__global__ __launch_bounds__(256) void upconv_kernel(
    const float* __restrict__ x, const _Float16* __restrict__ w1h,
    const float* __restrict__ b_up, const float* __restrict__ noise1,
    const float* __restrict__ ns1p, _Float16* __restrict__ h16)
{
    __shared__ __align__(64) _Float16 lds_in[17 * 17 * 32];  // [sp][ci]
    __shared__ __align__(64) _Float16 lds_w[4 * 32 * 32];    // [tap][co][ci]

    const int n   = blockIdx.z;
    const int cob = blockIdx.y >> 2;          // 0..5 -> *32 out channels
    const int par = blockIdx.y & 3;
    const int py  = par >> 1, px = par & 1;
    const int ti  = (blockIdx.x >> 3) * 16;   // parity-space tile origin (128/16=8)
    const int tj  = (blockIdx.x & 7) * 16;

    // deconv decomposition: taps valid for this output parity class
    const int nty = py ? 2 : 1, ntx = px ? 2 : 1;
    const int ntaps = nty * ntx;
    int tgt[4], tdi[4], tdj[4];
    for (int a = 0; a < nty; ++a)
        for (int b = 0; b < ntx; ++b) {
            int t = a * ntx + b;
            int ky = py ? (a * 2) : 1;
            int kx = px ? (b * 2) : 1;
            tgt[t] = ky * 3 + kx;
            tdi[t] = py ? a : 0;
            tdj[t] = px ? b : 0;
        }

    const int tid  = threadIdx.x;
    const int wave = tid >> 5;
    const int lane = tid & 31;
    const int half = lane >> 4;
    const int nidx = lane & 15;

    v8f acc[2][2] = {};   // [co_sub][px_sub]
    const float ns1 = ns1p[0];

    for (int cc = 0; cc < CIN; cc += 32) {
        __syncthreads();
        // stage input halo tile 17x17x32 (f32 -> f16, [sp][ci] transpose)
        for (int e = tid; e < 17 * 17 * 32; e += 256) {
            int sp = e % 289, ci = e / 289;
            int iy = sp / 17, ix = sp % 17;
            int gy = ti + iy, gx = tj + ix;
            float vv = 0.f;
            if (gy < HH && gx < WW)
                vv = x[((size_t)(n * CIN + cc + ci) * HH + gy) * WW + gx];
            lds_in[sp * 32 + ci] = (_Float16)vv;
        }
        // async-stage weights: ntaps x 32co x 32ci (layout-identical, f16)
        for (int i = tid; i < ntaps * 128; i += 256) {
            int f = i * 8;                    // f16 index in lds_w
            int ci8 = f & 31;
            int r = f >> 5;
            int co = r & 31, tp = r >> 5;
            const _Float16* g =
                &w1h[((size_t)(n * 9 + tgt[tp]) * CINT + cob * 32 + co) * CIN + cc + ci8];
            async_g2l_b128(&lds_w[f], g);
        }
        wait_async_all();
        __syncthreads();

        for (int tp = 0; tp < ntaps; ++tp) {
            v16h A[2], B[2];
#pragma unroll
            for (int pb = 0; pb < 2; ++pb) {
                int pp = wave * 32 + pb * 16 + nidx;            // pixel 0..255
                int ly = pp >> 4, lx = pp & 15;
                int sp = (ly + tdi[tp]) * 17 + (lx + tdj[tp]);
                B[pb] = *(const v16h*)&lds_in[sp * 32 + half * 16];
            }
#pragma unroll
            for (int cb = 0; cb < 2; ++cb) {
                const _Float16* wrow = &lds_w[tp * 1024 + (cb * 16 + nidx) * 32 + half * 16];
                v8h a0 = *(const v8h*)&wrow[0];
                v8h a1 = *(const v8h*)&wrow[8];
                A[cb] = __builtin_shufflevector(a0, a1, 0,1,2,3,4,5,6,7,8,9,10,11,12,13,14,15);
            }
#pragma unroll
            for (int cb = 0; cb < 2; ++cb)
#pragma unroll
                for (int pb = 0; pb < 2; ++pb)
                    acc[cb][pb] = __builtin_amdgcn_wmma_f32_16x16x32_f16(
                        false, A[cb], false, B[pb], (short)0, acc[cb][pb], false, false);
        }
    }

    // epilogue: + bias + noise, leaky-relu, f16 store (demod folded in weights)
#pragma unroll
    for (int pb = 0; pb < 2; ++pb) {
        int pp = wave * 32 + pb * 16 + nidx;
        int ly = pp >> 4, lx = pp & 15;
        int oy = 2 * (ti + ly) + py, ox = 2 * (tj + lx) + px;
        float nz = ns1 * noise1[(size_t)(n * H2 + oy) * W2 + ox];
#pragma unroll
        for (int cb = 0; cb < 2; ++cb)
#pragma unroll
            for (int r = 0; r < 8; ++r) {
                int co = cob * 32 + cb * 16 + r + half * 8;
                float vv = acc[cb][pb][r] + b_up[co];
                vv = lrelu(vv + nz);
                h16[((size_t)(n * CINT + co) * H2 + oy) * W2 + ox] = (_Float16)vv;
            }
    }
}

// ---------------------------------------------------------------------------
// Kernel 4: 3x3 conv 192->128 on 256x256; block 32co x 256px, 2x2 blocking.
// ---------------------------------------------------------------------------
__global__ __launch_bounds__(256) void conv2_kernel(
    const _Float16* __restrict__ h16, const _Float16* __restrict__ w2h,
    const float* __restrict__ b_conv, const float* __restrict__ noise2,
    const float* __restrict__ ns2p, float* __restrict__ h2)
{
    __shared__ __align__(64) _Float16 lds_in[18 * 18 * 32]; // halo [sp][ci]
    __shared__ __align__(64) _Float16 lds_w[9 * 32 * 32];   // [tap][co][ci]

    const int n   = blockIdx.z;
    const int cob = blockIdx.y;                // 0..3 -> *32 out channels
    const int ti  = (blockIdx.x >> 4) * 16;    // 256/16 = 16 tiles/dim
    const int tj  = (blockIdx.x & 15) * 16;

    const int tid  = threadIdx.x;
    const int wave = tid >> 5;
    const int lane = tid & 31;
    const int half = lane >> 4;
    const int nidx = lane & 15;

    v8f acc[2][2] = {};
    const float ns2 = ns2p[0];

    for (int cc = 0; cc < CINT; cc += 32) {
        __syncthreads();
        // stage input halo 18x18x32 (f16, [sp][ci] transpose, zero-pad border)
        for (int e = tid; e < 18 * 18 * 32; e += 256) {
            int sp = e % 324, ci = e / 324;
            int iy = sp / 18, ix = sp % 18;
            int gy = ti + iy - 1, gx = tj + ix - 1;
            _Float16 vv = (_Float16)0.f;
            if (gy >= 0 && gy < H2 && gx >= 0 && gx < W2)
                vv = h16[((size_t)(n * CINT + cc + ci) * H2 + gy) * W2 + gx];
            lds_in[sp * 32 + ci] = vv;
        }
        // async-stage weights: 9 x 32co x 32ci = 1152 x b128
        for (int i = tid; i < 1152; i += 256) {
            int f = i * 8;
            int ci8 = f & 31;
            int r = f >> 5;
            int co = r & 31, tp = r >> 5;
            const _Float16* g =
                &w2h[((size_t)(n * 9 + tp) * COUT + cob * 32 + co) * CINT + cc + ci8];
            async_g2l_b128(&lds_w[f], g);
        }
        wait_async_all();
        __syncthreads();

        for (int tp = 0; tp < 9; ++tp) {
            int dy = tp / 3, dx = tp % 3;
            v16h A[2], B[2];
#pragma unroll
            for (int pb = 0; pb < 2; ++pb) {
                int pp = wave * 32 + pb * 16 + nidx;
                int ly = pp >> 4, lx = pp & 15;
                int sp = (ly + dy) * 18 + (lx + dx);
                B[pb] = *(const v16h*)&lds_in[sp * 32 + half * 16];
            }
#pragma unroll
            for (int cb = 0; cb < 2; ++cb) {
                const _Float16* wrow = &lds_w[tp * 1024 + (cb * 16 + nidx) * 32 + half * 16];
                v8h a0 = *(const v8h*)&wrow[0];
                v8h a1 = *(const v8h*)&wrow[8];
                A[cb] = __builtin_shufflevector(a0, a1, 0,1,2,3,4,5,6,7,8,9,10,11,12,13,14,15);
            }
#pragma unroll
            for (int cb = 0; cb < 2; ++cb)
#pragma unroll
                for (int pb = 0; pb < 2; ++pb)
                    acc[cb][pb] = __builtin_amdgcn_wmma_f32_16x16x32_f16(
                        false, A[cb], false, B[pb], (short)0, acc[cb][pb], false, false);
        }
    }

#pragma unroll
    for (int pb = 0; pb < 2; ++pb) {
        int pp = wave * 32 + pb * 16 + nidx;
        int ly = pp >> 4, lx = pp & 15;
        int oy = ti + ly, ox = tj + lx;
        float nz = ns2 * noise2[(size_t)(n * H2 + oy) * W2 + ox];
#pragma unroll
        for (int cb = 0; cb < 2; ++cb)
#pragma unroll
            for (int r = 0; r < 8; ++r) {
                int co = cob * 32 + cb * 16 + r + half * 8;
                float vv = acc[cb][pb][r] + b_conv[co];
                vv = lrelu(vv + nz);
                h2[((size_t)(n * COUT + co) * H2 + oy) * W2 + ox] = vv;
            }
    }
}

// ---------------------------------------------------------------------------
// Kernel 5: RGB 1x1 conv (modulated, no demod) + bilinear-upsampled skip add.
// ---------------------------------------------------------------------------
__global__ void rgb_skip_kernel(const float* __restrict__ h2, const float* __restrict__ wrgb,
                                const float* __restrict__ b_rgb, const float* __restrict__ yin,
                                float* __restrict__ yout)
{
    int idx = blockIdx.x * blockDim.x + threadIdx.x; // NB*H2*W2
    int xq = idx % W2; int r = idx / W2;
    int yq = r % H2;   int n = r / H2;
    float a0 = 0.f, a1 = 0.f, a2 = 0.f;
    const float* wr = wrgb + n * IMGC * COUT;
    for (int co = 0; co < COUT; ++co) {
        float hv = h2[((size_t)(n * COUT + co) * H2 + yq) * W2 + xq];
        a0 += hv * wr[co];
        a1 += hv * wr[COUT + co];
        a2 += hv * wr[2 * COUT + co];
    }
    // half-pixel bilinear upsample 128->256 with edge clamp
    float fy = yq * 0.5f - 0.25f, fx = xq * 0.5f - 0.25f;
    int y0 = (int)floorf(fy); float wy = fy - (float)y0;
    int x0 = (int)floorf(fx); float wx = fx - (float)x0;
    int y0c = min(max(y0, 0), HH - 1), y1c = min(max(y0 + 1, 0), HH - 1);
    int x0c = min(max(x0, 0), WW - 1), x1c = min(max(x0 + 1, 0), WW - 1);
    float acc3[3] = {a0, a1, a2};
    for (int im = 0; im < IMGC; ++im) {
        const float* yp = yin + (size_t)(n * IMGC + im) * HH * WW;
        float up = (1.f - wy) * ((1.f - wx) * yp[y0c * WW + x0c] + wx * yp[y0c * WW + x1c])
                 +        wy  * ((1.f - wx) * yp[y1c * WW + x0c] + wx * yp[y1c * WW + x1c]);
        float rgb = acc3[im] + b_rgb[im];
        yout[((size_t)(n * IMGC + im) * H2 + yq) * W2 + xq] = up + lrelu(rgb);
    }
}

// ---------------------------------------------------------------------------
extern "C" void kernel_launch(void* const* d_in, const int* in_sizes, int n_in,
                              void* d_out, int out_size, void* d_ws, size_t ws_size,
                              hipStream_t stream)
{
    const float* x       = (const float*)d_in[0];
    const float* v       = (const float*)d_in[1];
    const float* y       = (const float*)d_in[2];
    const float* noise1  = (const float*)d_in[3];
    const float* noise2  = (const float*)d_in[4];
    const float* w_up    = (const float*)d_in[5];
    const float* b_up    = (const float*)d_in[6];
    const float* sw_up   = (const float*)d_in[7];
    const float* sb_up   = (const float*)d_in[8];
    const float* w_conv  = (const float*)d_in[9];
    const float* b_conv  = (const float*)d_in[10];
    const float* sw_conv = (const float*)d_in[11];
    const float* sb_conv = (const float*)d_in[12];
    const float* w_rgb   = (const float*)d_in[13];
    const float* b_rgb   = (const float*)d_in[14];
    const float* sw_rgb  = (const float*)d_in[15];
    const float* sb_rgb  = (const float*)d_in[16];
    const float* ns1     = (const float*)d_in[17];
    const float* ns2     = (const float*)d_in[18];

    char* ws = (char*)d_ws;
    auto alloc = [&](size_t bytes) -> char* {
        char* p = ws;
        ws += (bytes + 255) & ~(size_t)255;
        return p;
    };
    float*    s_up     = (float*)alloc((size_t)NB * CIN * 4);
    float*    sig_up   = (float*)alloc((size_t)NB * CINT * 4);
    float*    s_conv   = (float*)alloc((size_t)NB * CINT * 4);
    float*    sig_conv = (float*)alloc((size_t)NB * COUT * 4);
    float*    s_rgb    = (float*)alloc((size_t)NB * COUT * 4);
    float*    wrgb     = (float*)alloc((size_t)NB * IMGC * COUT * 4);
    _Float16* w1h      = (_Float16*)alloc((size_t)NB * 9 * CINT * CIN * 2);
    _Float16* w2h      = (_Float16*)alloc((size_t)NB * 9 * COUT * CINT * 2);
    _Float16* h16      = (_Float16*)alloc((size_t)NB * CINT * H2 * W2 * 2);

    float* h2   = (float*)d_out;
    float* yout = h2 + (size_t)NB * COUT * H2 * W2;

    styles_kernel<<<NB, 256, 0, stream>>>(v, sw_up, sb_up, w_up, sw_conv, sb_conv,
                                          w_conv, sw_rgb, sb_rgb, w_rgb,
                                          s_up, sig_up, s_conv, sig_conv, s_rgb, wrgb);
    {
        int tot = NB * 9 * CINT * CIN;
        modw_up_kernel<<<(tot + 255) / 256, 256, 0, stream>>>(w_up, s_up, sig_up, w1h);
    }
    {
        int tot = NB * 9 * COUT * CINT;
        modw_conv_kernel<<<(tot + 255) / 256, 256, 0, stream>>>(w_conv, s_conv, sig_conv, w2h);
    }
    {
        dim3 g(64, 24, NB);  // 8x8 parity tiles of 16x16, (6 co-blocks x 4 parities), N
        upconv_kernel<<<g, 256, 0, stream>>>(x, w1h, b_up, noise1, ns1, h16);
    }
    {
        dim3 g(256, 4, NB);  // 16x16 tiles of 16x16, 4 co-blocks, N
        conv2_kernel<<<g, 256, 0, stream>>>(h16, w2h, b_conv, noise2, ns2, h2);
    }
    rgb_skip_kernel<<<(NB * H2 * W2) / 256, 256, 0, stream>>>(h2, wrgb, b_rgb, y, yout);
    (void)in_sizes; (void)n_in; (void)out_size; (void)ws_size;
}
